// MultiHeadSelfAttention_64132451663943
// MI455X (gfx1250) — matrix operand order repaired and checked by reading, not compiled, not run on hardware
//
#include <hip/hip_runtime.h>

typedef unsigned short u16;
typedef unsigned int   u32;
typedef __attribute__((ext_vector_type(16))) __bf16 v16bf;
typedef __attribute__((ext_vector_type(8)))  float  v8f;

static __device__ __forceinline__ u16 f2bf(float f) {
    u32 u = __float_as_uint(f);
    u32 r = u + 0x7FFFu + ((u >> 16) & 1u);   // round-to-nearest-even
    return (u16)(r >> 16);
}
static __device__ __forceinline__ float bf2f(u16 h) {
    return __uint_as_float(((u32)h) << 16);
}

// ---------------------------------------------------------------------------
// Async global->LDS (CDNA5): builtin takes v4i pointers in addrspace(1)/(3).
// Guarded so a missing builtin falls back to register-staged copies.
// ---------------------------------------------------------------------------
#if defined(__AMDGCN__) && __has_builtin(__builtin_amdgcn_global_load_async_to_lds_b128) && __has_builtin(__builtin_amdgcn_s_wait_asynccnt)
#define USE_ASYNC_LDS 1
#else
#define USE_ASYNC_LDS 0
#endif

#if USE_ASYNC_LDS
typedef int v4i __attribute__((vector_size(16)));
typedef __attribute__((address_space(1))) v4i as1_v4i;
typedef __attribute__((address_space(3))) v4i as3_v4i;
static __device__ __forceinline__ void async_copy_b128(void* lds_dst, const void* gsrc) {
    __builtin_amdgcn_global_load_async_to_lds_b128(
        (as1_v4i*)(void*)gsrc, (as3_v4i*)lds_dst, 0, 0);
}
#endif

// ---------------------------------------------------------------------------
// Wt[n][k] = (bf16) W[k][n] via 32x32 LDS tile: coalesced reads AND writes.
// ---------------------------------------------------------------------------
__global__ __launch_bounds__(256)
void transpose_convert_kernel(const float* __restrict__ W, u16* __restrict__ Wt,
                              int K, int N) {
    __shared__ u16 tile[32][33];
    const int bk = blockIdx.y * 32, bn = blockIdx.x * 32;
    const int tx = threadIdx.x & 31, ty = threadIdx.x >> 5;   // 32 x 8
    #pragma unroll
    for (int i = 0; i < 32; i += 8)
        tile[ty + i][tx] = f2bf(W[(size_t)(bk + ty + i) * N + bn + tx]);
    __syncthreads();
    #pragma unroll
    for (int i = 0; i < 32; i += 8)
        Wt[(size_t)(bn + ty + i) * K + bk + tx] = tile[tx][ty + i];
}

// ---------------------------------------------------------------------------
// Tiled bf16 WMMA GEMM:  Out[M,N] = A[M,K] * Bt[N,K]^T + bias[N]
// 128x128 block tile, BK=32, 8 waves (4Mx2N), 8 wmma / k-step.
// Double-buffered LDS; bf16 tiles streamed with global_load_async_to_lds_b128
// (ASYNCcnt) when available; fp32 A converted to bf16 through registers.
// ---------------------------------------------------------------------------
#define BM 128
#define BN 128
#define BK 32

static __device__ __forceinline__ v16bf make_frag(const u16* p0, const u16* p1) {
    union { uint4 u[2]; v16bf v; } t;
    t.u[0] = *(const uint4*)p0;   // ds_load_b128
    t.u[1] = *(const uint4*)p1;   // ds_load_b128
    return t.v;
}

template <bool A_F32, bool OUT_F32>
__global__ __launch_bounds__(256)
void gemm_wmma_kernel(const void* __restrict__ Aptr, const u16* __restrict__ Bt,
                      const float* __restrict__ bias, void* __restrict__ Out,
                      int M, int N, int K) {
    __shared__ __attribute__((aligned(32))) u16 As[2][BM * BK];
    __shared__ __attribute__((aligned(32))) u16 Bs[2][BN * BK];

    const int tid    = threadIdx.x;
    const int wave   = tid >> 5;
    const int lane   = tid & 31;
    const int hlf    = lane >> 4;          // lane half (wave32 WMMA layout)
    const int r      = lane & 15;
    const int wave_m = wave & 3;           // 4 waves along M (32 rows each)
    const int wave_n = wave >> 2;          // 2 waves along N (64 cols each)
    const int bm     = blockIdx.y * BM;
    const int bn     = blockIdx.x * BN;

    const int srow    = tid >> 1;          // staging: 2 threads per row
    const int shalf   = tid & 1;           // each stages 16 elements (32B)
    const int lds_off = srow * BK + shalf * 16;

    const float* Af = (const float*)Aptr;
    const u16*   Ah = (const u16*)Aptr;
    const size_t a_off = (size_t)(bm + srow) * K + shalf * 16;
    const size_t b_off = (size_t)(bn + srow) * K + shalf * 16;

    v8f acc[2][4];
    #pragma unroll
    for (int i = 0; i < 2; i++)
        #pragma unroll
        for (int j = 0; j < 4; j++)
            #pragma unroll
            for (int e = 0; e < 8; e++) acc[i][j][e] = 0.0f;

    // ---------------- prologue: stage k-step 0 into buffer 0 ----------------
    if (A_F32) {
        const float4* s4 = (const float4*)(Af + a_off);
        union { float4 v4[4]; float f[16]; } a;
        #pragma unroll
        for (int i = 0; i < 4; i++) a.v4[i] = s4[i];
        union { u16 h[16]; uint4 q[2]; } t;
        #pragma unroll
        for (int i = 0; i < 16; i++) t.h[i] = f2bf(a.f[i]);
        uint4* d = (uint4*)&As[0][lds_off];
        d[0] = t.q[0]; d[1] = t.q[1];
    } else {
#if USE_ASYNC_LDS
        async_copy_b128(&As[0][lds_off],     Ah + a_off);
        async_copy_b128(&As[0][lds_off + 8], Ah + a_off + 8);
#else
        const uint4* sa = (const uint4*)(Ah + a_off);
        uint4* d = (uint4*)&As[0][lds_off];
        d[0] = sa[0]; d[1] = sa[1];
#endif
    }
    {
#if USE_ASYNC_LDS
        async_copy_b128(&Bs[0][lds_off],     Bt + b_off);
        async_copy_b128(&Bs[0][lds_off + 8], Bt + b_off + 8);
#else
        const uint4* sb = (const uint4*)(Bt + b_off);
        uint4* d = (uint4*)&Bs[0][lds_off];
        d[0] = sb[0]; d[1] = sb[1];
#endif
    }
#if USE_ASYNC_LDS
    __builtin_amdgcn_s_wait_asynccnt(0);
#endif
    __syncthreads();

    // ---------------- main loop: 1 barrier per k-step ----------------
    const int nsteps = K / BK;
    for (int st = 0; st < nsteps; st++) {
        const int  cur  = st & 1, nxt = cur ^ 1;
        const bool pref = (st + 1) < nsteps;
        const int  ktn  = (st + 1) * BK;

        // issue next-tile prefetch (async straight to LDS, or to registers)
        union { float4 v4[4]; float f[16]; } apf;
        uint4 aq0, aq1, bq0, bq1;
        if (pref) {
            if (A_F32) {
                const float4* s4 = (const float4*)(Af + a_off + ktn);
                #pragma unroll
                for (int i = 0; i < 4; i++) apf.v4[i] = s4[i];
            } else {
#if USE_ASYNC_LDS
                async_copy_b128(&As[nxt][lds_off],     Ah + a_off + ktn);
                async_copy_b128(&As[nxt][lds_off + 8], Ah + a_off + ktn + 8);
#else
                const uint4* sa = (const uint4*)(Ah + a_off + ktn);
                aq0 = sa[0]; aq1 = sa[1];
#endif
            }
#if USE_ASYNC_LDS
            async_copy_b128(&Bs[nxt][lds_off],     Bt + b_off + ktn);
            async_copy_b128(&Bs[nxt][lds_off + 8], Bt + b_off + ktn + 8);
#else
            const uint4* sb = (const uint4*)(Bt + b_off + ktn);
            bq0 = sb[0]; bq1 = sb[1];
#endif
        }

        // ---- fragments from current buffer ----
        // A: lanes 0-15 hold K{0..7,16..23}, lanes 16-31 hold K{8..15,24..31}
        v16bf afrag[2], bfrag[4];
        #pragma unroll
        for (int mf = 0; mf < 2; mf++) {
            int row = wave_m * 32 + mf * 16 + r;
            afrag[mf] = make_frag(&As[cur][row * BK + hlf * 8],
                                  &As[cur][row * BK + 16 + hlf * 8]);
        }
        #pragma unroll
        for (int nf = 0; nf < 4; nf++) {
            int col = wave_n * 64 + nf * 16 + r;
            bfrag[nf] = make_frag(&Bs[cur][col * BK + hlf * 8],
                                  &Bs[cur][col * BK + 16 + hlf * 8]);
        }

        #pragma unroll
        for (int mf = 0; mf < 2; mf++)
            #pragma unroll
            for (int nf = 0; nf < 4; nf++)
                acc[mf][nf] = __builtin_amdgcn_wmma_f32_16x16x32_bf16(
                    false, afrag[mf], false, bfrag[nf],
                    (short)0, acc[mf][nf], false, false);

        // ---- commit register prefetch into next buffer ----
        if (pref) {
            if (A_F32) {
                union { u16 h[16]; uint4 q[2]; } t;
                #pragma unroll
                for (int i = 0; i < 16; i++) t.h[i] = f2bf(apf.f[i]);
                uint4* d = (uint4*)&As[nxt][lds_off];
                d[0] = t.q[0]; d[1] = t.q[1];
            }
#if !USE_ASYNC_LDS
            if (!A_F32) {
                uint4* d = (uint4*)&As[nxt][lds_off];
                d[0] = aq0; d[1] = aq1;
            }
            {
                uint4* d = (uint4*)&Bs[nxt][lds_off];
                d[0] = bq0; d[1] = bq1;
            }
#endif
        }
#if USE_ASYNC_LDS
        __builtin_amdgcn_s_wait_asynccnt(0);
#endif
        __syncthreads();
    }

    // ---- epilogue: C layout VGPR j -> (M=j or j+8, N=lane%16) ----
    #pragma unroll
    for (int mf = 0; mf < 2; mf++) {
        #pragma unroll
        for (int nf = 0; nf < 4; nf++) {
            int col = bn + wave_n * 64 + nf * 16 + r;
            float bv = bias[col];
            #pragma unroll
            for (int j = 0; j < 8; j++) {
                int row = bm + wave_m * 32 + mf * 16 + hlf * 8 + j;
                float v = acc[mf][nf][j] + bv;
                if (OUT_F32) ((float*)Out)[(size_t)row * N + col] = v;
                else         ((u16*)Out)[(size_t)row * N + col]  = f2bf(v);
            }
        }
    }
}

// ---------------------------------------------------------------------------
// Per-token 16-head attention (softmax over heads, width 16 — pure VALU).
// qkv row layout per token: col = head*192 + {0..63 q | 64..127 k | 128..191 v}
// Each lane owns one (token, head); k/v tiles staged in LDS (broadcast reads).
// ---------------------------------------------------------------------------
#define TOK_PER_BLOCK 16

__global__ __launch_bounds__(256)
void attn_kernel(const u16* __restrict__ qkv, u16* __restrict__ ctx) {
    __shared__ __attribute__((aligned(16))) u16 ks[TOK_PER_BLOCK][16][64];
    __shared__ __attribute__((aligned(16))) u16 vs[TOK_PER_BLOCK][16][64];

    const int tid = threadIdx.x;
    const size_t tok0 = (size_t)blockIdx.x * TOK_PER_BLOCK;

    // stage k,v for 16 tokens: 4096 16B chunks, 16 per thread
    #pragma unroll
    for (int i = 0; i < 16; i++) {
        int c      = tid + i * 256;
        int tg     = c >> 4;
        int within = c & 15;
        int t   = tg >> 4, g = tg & 15;
        int sel = within >> 3;          // 0=k, 1=v
        int prt = within & 7;
        const uint4* src = (const uint4*)(qkv + (tok0 + t) * 3072 + g * 192
                                          + 64 + sel * 64 + prt * 8);
        uint4* dst = (uint4*)((sel ? &vs[t][g][0] : &ks[t][g][0]) + prt * 8);
        *dst = *src;
    }
    __syncthreads();

    const int wave = tid >> 5, lane = tid & 31;
    const int tl = wave * 2 + (lane >> 4);   // local token (2 per wave)
    const int h  = lane & 15;                // head owned by this lane
    const size_t t = tok0 + tl;

    // load q_h into registers
    float buf[64];
    {
        const uint4* qsrc = (const uint4*)(qkv + t * 3072 + h * 192);
        #pragma unroll
        for (int i = 0; i < 8; i++) {
            uint4 q4 = qsrc[i];
            u32 w[4] = { q4.x, q4.y, q4.z, q4.w };
            #pragma unroll
            for (int j = 0; j < 4; j++) {
                buf[i * 8 + j * 2]     = bf2f((u16)(w[j] & 0xFFFFu));
                buf[i * 8 + j * 2 + 1] = bf2f((u16)(w[j] >> 16));
            }
        }
    }

    // scores s[g] = (q_h . k_g) / sqrt(64)
    float s[16];
    #pragma unroll
    for (int g = 0; g < 16; g++) {
        float a = 0.0f;
        const u32* kp = (const u32*)&ks[tl][g][0];
        #pragma unroll
        for (int d2 = 0; d2 < 32; d2++) {
            u32 w = kp[d2];
            a += buf[2 * d2]     * bf2f((u16)(w & 0xFFFFu));
            a += buf[2 * d2 + 1] * bf2f((u16)(w >> 16));
        }
        s[g] = a * 0.125f;
    }

    // softmax over the 16 heads (entirely in-lane)
    float m = s[0];
    #pragma unroll
    for (int g = 1; g < 16; g++) m = fmaxf(m, s[g]);
    float e[16], sum = 0.0f;
    #pragma unroll
    for (int g = 0; g < 16; g++) { e[g] = __expf(s[g] - m); sum += e[g]; }
    const float inv = 1.0f / sum;

    // context row: reuse buf as the accumulator
    #pragma unroll
    for (int d = 0; d < 64; d++) buf[d] = 0.0f;
    #pragma unroll
    for (int g = 0; g < 16; g++) {
        float wgt = e[g];
        const u32* vp = (const u32*)&vs[tl][g][0];
        #pragma unroll
        for (int d2 = 0; d2 < 32; d2++) {
            u32 w = vp[d2];
            buf[2 * d2]     += wgt * bf2f((u16)(w & 0xFFFFu));
            buf[2 * d2 + 1] += wgt * bf2f((u16)(w >> 16));
        }
    }

    union { u16 h16[64]; uint4 q[8]; } o;
    #pragma unroll
    for (int d = 0; d < 64; d++) o.h16[d] = f2bf(buf[d] * inv);
    uint4* dst = (uint4*)(ctx + t * 1024 + h * 64);
    #pragma unroll
    for (int i = 0; i < 8; i++) dst[i] = o.q[i];
}

// ---------------------------------------------------------------------------
extern "C" void kernel_launch(void* const* d_in, const int* in_sizes, int n_in,
                              void* d_out, int out_size, void* d_ws, size_t ws_size,
                              hipStream_t stream) {
    const float* x     = (const float*)d_in[0];   // [2,8192,1024]
    const float* W_qkv = (const float*)d_in[1];   // [1024,3072]
    const float* b_qkv = (const float*)d_in[2];   // [3072]
    const float* W_out = (const float*)d_in[3];   // [1024,1024]
    const float* b_out = (const float*)d_in[4];   // [1024]
    float* out = (float*)d_out;                   // [2,8192,1024] fp32

    const int M  = 2 * 8192;   // tokens
    const int D  = 1024;
    const int N1 = 3 * 1024;

    // workspace layout (bf16): Wt_qkv 6MB | Wt_out 2MB | qkv 96MB | ctx 32MB
    char* ws = (char*)d_ws;
    u16* Wtq = (u16*)ws; ws += (size_t)N1 * D * sizeof(u16);
    u16* Wto = (u16*)ws; ws += (size_t)D * D * sizeof(u16);
    u16* qkv = (u16*)ws; ws += (size_t)M * N1 * sizeof(u16);
    u16* ctx = (u16*)ws;

    transpose_convert_kernel<<<dim3(N1 / 32, D / 32), 256, 0, stream>>>(W_qkv, Wtq, D, N1);
    transpose_convert_kernel<<<dim3(D / 32, D / 32), 256, 0, stream>>>(W_out, Wto, D, D);

    gemm_wmma_kernel<true, false><<<dim3(N1 / BN, M / BM), 256, 0, stream>>>(
        x, Wtq, b_qkv, qkv, M, N1, D);

    attn_kernel<<<M / TOK_PER_BLOCK, 256, 0, stream>>>(qkv, ctx);

    gemm_wmma_kernel<false, true><<<dim3(D / BN, M / BM), 256, 0, stream>>>(
        ctx, Wto, b_out, out, M, D, D);
}
